// ROUNDModel_9861244912250
// MI455X (gfx1250) — compile-verified, matrix operand order checked by reading, not run on hardware
//
#include <hip/hip_runtime.h>

typedef __attribute__((ext_vector_type(16))) _Float16 v16h;
typedef __attribute__((ext_vector_type(8)))  float    v8f;

#define HID    1024
#define IN_DIM 512
#define BATCH  4096

// Fragment element container: 16 halves = 8 dwords = one WMMA A/B operand per lane.
union AFrag { v16h v; _Float16 h[16]; unsigned int d[8]; };

// ---------------------------------------------------------------------------
// Kernel 0: pre-swizzle W_e (512x1024) and W_d (4096x1024) f32 row-major into
// f16 WMMA B-operand fragments. Fragment (kt,nt) covers K=kt*32..+31,
// N=nt*16..+15. Per-lane layout follows the CDNA5 16-bit operand map:
// lane = {group=lane>>4, idx=lane&15}; half h -> v=h/2;
// k_local = (v<4 ? 2v : 16+2(v-4)) + (h&1) + 8*group ; N (or M) = idx.
// ---------------------------------------------------------------------------
__global__ void __launch_bounds__(256) prep_frags(const float* __restrict__ We,
                                                  const float* __restrict__ Wd,
                                                  v16h* __restrict__ weF,
                                                  v16h* __restrict__ wdF) {
  const int WE_FRAGS = (IN_DIM / 32) * (HID / 16);    // 16*64  = 1024
  const int WD_FRAGS = ((4 * HID) / 32) * (HID / 16); // 128*64 = 8192
  int gid = blockIdx.x * 256 + threadIdx.x;
  if (gid >= (WE_FRAGS + WD_FRAGS) * 32) return;
  int lane = gid & 31, frag = gid >> 5;
  int group = lane >> 4, idx = lane & 15;
  const float* W;
  v16h* dst;
  int fi;
  if (frag < WE_FRAGS) { W = We; dst = weF; fi = frag; }
  else                 { W = Wd; dst = wdF; fi = frag - WE_FRAGS; }
  int kt = fi >> 6, nt = fi & 63;
  AFrag u;
#pragma unroll
  for (int h = 0; h < 16; ++h) {
    int v = h >> 1;
    int kl = ((v < 4) ? 2 * v : 16 + 2 * (v - 4)) + (h & 1) + group * 8;
    u.h[h] = (_Float16)W[(size_t)(kt * 32 + kl) * HID + nt * 16 + idx];
  }
  dst[(size_t)fi * 32 + lane] = u.v;
}

// ---------------------------------------------------------------------------
// Kernel 1: embed. p = x @ W_e + b_e for a 16-row batch tile (WMMA, K=512),
// then write cos(p), sin(p) as f16 row-major to workspace.
// ---------------------------------------------------------------------------
__global__ void __launch_bounds__(256) embed_kernel(const float* __restrict__ x,
                                                    const float* __restrict__ be,
                                                    const v16h* __restrict__ weF,
                                                    _Float16* __restrict__ xpc,
                                                    _Float16* __restrict__ xps) {
  extern __shared__ char smem[];
  float* xs = (float*)smem; // [16][512] f32 x tile
  int bt = blockIdx.x, tid = threadIdx.x;

  const float4* src = (const float4*)(x + (size_t)bt * 16 * IN_DIM);
  float4* dst4 = (float4*)xs;
  for (int i = tid; i < 16 * IN_DIM / 4; i += 256) dst4[i] = src[i];
  __syncthreads();

  int wave = tid >> 5, lane = tid & 31, group = lane >> 4, idx = lane & 15;
  int n0 = wave * 128;

  v8f acc[8];
#pragma unroll
  for (int f = 0; f < 8; ++f) {
    float b = be[n0 + 16 * f + idx];
#pragma unroll
    for (int r = 0; r < 8; ++r) acc[f][r] = b;
  }

  for (int kt = 0; kt < IN_DIM / 32; ++kt) {
    AFrag au;
    const float* rb = xs + idx * IN_DIM + kt * 32 + group * 8;
#pragma unroll
    for (int j = 0; j < 8; ++j) {
      int pb = (j < 4) ? 2 * j : 16 + 2 * (j - 4);
      au.h[2 * j]     = (_Float16)rb[pb];
      au.h[2 * j + 1] = (_Float16)rb[pb + 1];
    }
    const v16h* wb = weF + (size_t)(kt * 64 + wave * 8) * 32 + lane;
#pragma unroll
    for (int f = 0; f < 8; ++f) {
      acc[f] = __builtin_amdgcn_wmma_f32_16x16x32_f16(
          false, au.v, false, wb[f * 32], (short)0, acc[f], false, false);
    }
  }

#pragma unroll
  for (int f = 0; f < 8; ++f) {
#pragma unroll
    for (int r = 0; r < 8; ++r) {
      float p = acc[f][r];
      size_t o = (size_t)(bt * 16 + r + 8 * group) * HID + (n0 + 16 * f + idx);
      xpc[o] = (_Float16)__cosf(p);
      xps[o] = (_Float16)__sinf(p);
    }
  }
}

// ---------------------------------------------------------------------------
// Kernel 2: 12-step recurrence. 32 batch rows per workgroup, 512 threads
// (16 waves); each wave owns 4 N-tiles x 2 M-tiles -> every W_d fragment
// loaded from L2 feeds 2 WMMAs (halves L2 traffic vs 16-row tiles).
// ph lives in the WMMA accumulators. LDS inp tile [32][4096] f16 (256 KB):
// cols 0..2047 = cos(ph),sin(ph) rewritten each step; cols 2048..4095 =
// cos(p),sin(p) filled once.
// ---------------------------------------------------------------------------
__global__ void __launch_bounds__(512) osc_kernel(const v16h* __restrict__ wdF,
                                                  const _Float16* __restrict__ xpc,
                                                  const _Float16* __restrict__ xps,
                                                  const float* __restrict__ bd_g,
                                                  const float* __restrict__ Wr,
                                                  const float* __restrict__ br,
                                                  const int* __restrict__ steps_p,
                                                  float* __restrict__ out,
                                                  float* __restrict__ H) {
  extern __shared__ char smem[];
  _Float16* inp = (_Float16*)smem;          // [32][4096] halves
  unsigned int* inpU = (unsigned int*)smem; // same, as dword pairs
  float* red = (float*)(smem + 32 * 4096 * 2);

  int bt = blockIdx.x, tid = threadIdx.x;
  int wave = tid >> 5, lane = tid & 31, group = lane >> 4, idx = lane & 15;
  int n0 = wave * 64; // 4 N-tiles of 16 per wave
  int b0 = bt * 32;
  int steps = *steps_p;

  // Fill xp half of the inp tile once (uints 1024..2047 of each 2048-uint row).
  const unsigned int* xcU = (const unsigned int*)xpc;
  const unsigned int* xsU = (const unsigned int*)xps;
  for (int i = tid; i < 32 * 512; i += 512) {
    int m = i >> 9, c = i & 511;
    inpU[m * 2048 + 1024 + c] = xcU[(size_t)(b0 + m) * 512 + c];
    inpU[m * 2048 + 1536 + c] = xsU[(size_t)(b0 + m) * 512 + c];
  }

  float bd[4];
#pragma unroll
  for (int f = 0; f < 4; ++f) bd[f] = bd_g[n0 + 16 * f + idx];

  v8f acc[2][4] = {}; // [mtile][ntile], ph0 = 0

  for (int t = 0; t < steps; ++t) {
    // cos/sin of current phases -> LDS (each wave fills its 64 cols x 32 rows)
#pragma unroll
    for (int mt = 0; mt < 2; ++mt) {
#pragma unroll
      for (int f = 0; f < 4; ++f) {
#pragma unroll
        for (int r = 0; r < 8; ++r) {
          float p = acc[mt][f][r];
          int m = mt * 16 + r + 8 * group, n = n0 + 16 * f + idx;
          inp[m * 4096 + n]        = (_Float16)__cosf(p);
          inp[m * 4096 + 1024 + n] = (_Float16)__sinf(p);
        }
      }
    }
    __syncthreads();

    for (int kt = 0; kt < 128; ++kt) {
      AFrag a0, a1;
      const uint4* r0 = (const uint4*)(inpU + idx * 2048 + kt * 16 + group * 4);
      const uint4* r1 = (const uint4*)(inpU + (16 + idx) * 2048 + kt * 16 + group * 4);
      uint4 q;
      q = r0[0]; a0.d[0] = q.x; a0.d[1] = q.y; a0.d[2] = q.z; a0.d[3] = q.w;
      q = r0[2]; a0.d[4] = q.x; a0.d[5] = q.y; a0.d[6] = q.z; a0.d[7] = q.w;
      q = r1[0]; a1.d[0] = q.x; a1.d[1] = q.y; a1.d[2] = q.z; a1.d[3] = q.w;
      q = r1[2]; a1.d[4] = q.x; a1.d[5] = q.y; a1.d[6] = q.z; a1.d[7] = q.w;
      const v16h* wb = wdF + (size_t)(kt * 64 + wave * 4) * 32 + lane;
#pragma unroll
      for (int f = 0; f < 4; ++f) {
        v16h b = wb[f * 32];
        acc[0][f] = __builtin_amdgcn_wmma_f32_16x16x32_f16(
            false, a0.v, false, b, (short)0, acc[0][f], false, false);
        acc[1][f] = __builtin_amdgcn_wmma_f32_16x16x32_f16(
            false, a1.v, false, b, (short)0, acc[1][f], false, false);
      }
    }

#pragma unroll
    for (int mt = 0; mt < 2; ++mt)
#pragma unroll
      for (int f = 0; f < 4; ++f)
#pragma unroll
        for (int r = 0; r < 8; ++r) acc[mt][f][r] += bd[f];

    // H[t] = ph
    float* Ht = H + (size_t)t * BATCH * HID;
#pragma unroll
    for (int mt = 0; mt < 2; ++mt) {
#pragma unroll
      for (int f = 0; f < 4; ++f) {
#pragma unroll
        for (int r = 0; r < 8; ++r) {
          Ht[(size_t)(b0 + mt * 16 + r + 8 * group) * HID + (n0 + 16 * f + idx)] =
              acc[mt][f][r];
        }
      }
    }
    __syncthreads(); // K-loop reads done before next step's LDS rewrite
  }

  // out = [cos ph, sin ph, ph] @ W_r + b_r, reduced across the 32-row tile.
  if (tid < 32) red[tid] = 0.f;
  __syncthreads();
  float pr[2][8] = {};
#pragma unroll
  for (int f = 0; f < 4; ++f) {
    int n = n0 + 16 * f + idx;
    float w0 = Wr[n], w1 = Wr[HID + n], w2 = Wr[2 * HID + n];
#pragma unroll
    for (int mt = 0; mt < 2; ++mt) {
#pragma unroll
      for (int r = 0; r < 8; ++r) {
        float p = acc[mt][f][r];
        pr[mt][r] += __cosf(p) * w0 + __sinf(p) * w1 + p * w2;
      }
    }
  }
#pragma unroll
  for (int mt = 0; mt < 2; ++mt)
#pragma unroll
    for (int r = 0; r < 8; ++r)
      atomicAdd(&red[mt * 16 + r + 8 * group], pr[mt][r]);
  __syncthreads();
  if (tid < 32) out[b0 + tid] = red[tid] + br[0];
}

// ---------------------------------------------------------------------------
extern "C" void kernel_launch(void* const* d_in, const int* in_sizes, int n_in,
                              void* d_out, int out_size, void* d_ws, size_t ws_size,
                              hipStream_t stream) {
  const float* x  = (const float*)d_in[0];
  const float* We = (const float*)d_in[1];
  const float* be = (const float*)d_in[2];
  const float* Wd = (const float*)d_in[3];
  const float* bd = (const float*)d_in[4];
  const float* Wr = (const float*)d_in[5];
  const float* br = (const float*)d_in[6];
  const int*   st = (const int*)d_in[7];

  char* ws = (char*)d_ws;
  v16h*     weF = (v16h*)ws;                          // 1 MB of W_e frags
  v16h*     wdF = (v16h*)(ws + (size_t)(1u << 20));   // 8 MB of W_d frags
  _Float16* xpc = (_Float16*)(ws + (size_t)(9u << 20));  // 8 MB cos(p)
  _Float16* xps = (_Float16*)(ws + (size_t)(17u << 20)); // 8 MB sin(p)

  float* out = (float*)d_out;
  float* H   = out + BATCH;

  prep_frags<<<1152, 256, 0, stream>>>(We, Wd, weF, wdF);
  embed_kernel<<<BATCH / 16, 256, 16 * IN_DIM * 4, stream>>>(x, be, weF, xpc, xps);
  osc_kernel<<<BATCH / 32, 512, 32 * 4096 * 2 + 128, stream>>>(
      wdF, xpc, xps, bd, Wr, br, st, out, H);
}
